// MLPLinkPredictor_10685878632451
// MI455X (gfx1250) — matrix operand order repaired and checked by reading, not compiled, not run on hardware
//
#include <hip/hip_runtime.h>
#include <hip/hip_bf16.h>

typedef __attribute__((ext_vector_type(16))) __bf16 v16bf;
typedef __attribute__((ext_vector_type(8)))  float  v8f;

#define N_NODES 100000
#define N_EDGES 1000000
#define IN_C    256
#define HID     64
#define C1      128   // 2*HID

__device__ __forceinline__ __bf16 f2bf(float f) { return (__bf16)f; }

// ---- WMMA fragment loaders (CDNA5 16x16x32 bf16 layouts, wave32) ----
// A: 16x32 tile from row-major fp32 (lda = row stride)
__device__ __forceinline__ v16bf load_a_frag(const float* __restrict__ A, int lda,
                                             int m0, int k0, int lane) {
  int m    = m0 + (lane & 15);
  int koff = (lane & 16) ? 8 : 0;
  const float* row = A + (size_t)m * lda + k0;
  v16bf a;
#pragma unroll
  for (int v = 0; v < 4; ++v) {
    a[2*v]     = f2bf(row[koff + 2*v]);
    a[2*v+1]   = f2bf(row[koff + 2*v + 1]);
    a[8+2*v]   = f2bf(row[16 + koff + 2*v]);
    a[8+2*v+1] = f2bf(row[16 + koff + 2*v + 1]);
  }
  return a;
}

// B: 32x16 tile from row-major fp32 [K, N] (used only by the one-time pack kernels)
__device__ __forceinline__ v16bf load_b_frag(const float* __restrict__ B, int ldb,
                                             int k0, int n0, int lane) {
  int n    = n0 + (lane & 15);
  int koff = (lane & 16) ? 8 : 0;
  const float* col = B + (size_t)(k0 + koff) * ldb + n;
  v16bf b;
#pragma unroll
  for (int j = 0; j < 4; ++j) {
    b[2*j]     = f2bf(col[(2*j)      * (size_t)ldb]);
    b[2*j+1]   = f2bf(col[(2*j + 1)  * (size_t)ldb]);
    b[8+2*j]   = f2bf(col[(16 + 2*j)     * (size_t)ldb]);
    b[8+2*j+1] = f2bf(col[(16 + 2*j + 1) * (size_t)ldb]);
  }
  return b;
}

// ---- one-time weight pre-pack: fp32 [K,N] -> bf16 per-lane WMMA fragments ----
// layout: P[((kt * (N/16) + nt) * 32 + lane) * 16 + elem], 32B per lane => b128 loads
__global__ void pack_b_kernel(const float* __restrict__ B, __bf16* __restrict__ P,
                              int Ncols) {
  int ntn  = Ncols / 16;
  int tile = blockIdx.x;
  int kt   = tile / ntn, nt = tile % ntn;
  int lane = threadIdx.x & 31;
  v16bf b = load_b_frag(B, Ncols, kt * 32, nt * 16, lane);
  *(v16bf*)(P + ((size_t)tile * 32 + lane) * 16) = b;
}

// ---- dense GEMM: C[M,Ncols] = A[M,KDIM] @ B, one 16x16 tile per wave ----
template <int KDIM>
__global__ void gemm_bf16_kernel(const float* __restrict__ A,
                                 const __bf16* __restrict__ Pb,
                                 float* __restrict__ C, int Ncols) {
  int wave = threadIdx.x >> 5;
  int lane = threadIdx.x & 31;
  int m0   = blockIdx.x * 16;
  int n0   = wave * 16;
  int ntn  = Ncols / 16;
  v8f acc = {};
#pragma unroll
  for (int k0 = 0; k0 < KDIM; k0 += 32) {
    v16bf a = load_a_frag(A, KDIM, m0, k0, lane);
    int tile = (k0 / 32) * ntn + wave;
    v16bf b  = *(const v16bf*)(Pb + ((size_t)tile * 32 + lane) * 16);
    acc = __builtin_amdgcn_wmma_f32_16x16x32_bf16(false, a, false, b, (short)0, acc,
                                                  false, false);
  }
  int n     = n0 + (lane & 15);
  int mbase = m0 + ((lane & 16) ? 8 : 0);
#pragma unroll
  for (int r = 0; r < 8; ++r)
    C[(size_t)(mbase + r) * Ncols + n] = acc[r];
}

// ---- degree / norm kernels ----
__global__ void deg_init(float* __restrict__ deg) {
  int i = blockIdx.x * 256 + threadIdx.x;
  if (i < N_NODES) deg[i] = 1.0f;  // self-loop
}
__global__ void deg_count(const int* __restrict__ dst, float* __restrict__ deg) {
  int e = blockIdx.x * 256 + threadIdx.x;
  if (e < N_EDGES) atomicAdd(&deg[dst[e]], 1.0f);
}
__global__ void deg_rsqrt(float* __restrict__ deg) {
  int i = blockIdx.x * 256 + threadIdx.x;
  if (i < N_NODES) deg[i] = rsqrtf(deg[i]);
}

// ---- GCN aggregation ----
__global__ void selfloop_init(const float* __restrict__ XW, const float* __restrict__ dinv,
                              float* __restrict__ H, int channels) {
  int i = blockIdx.x, c = threadIdx.x;
  float dv = dinv[i];
  H[(size_t)i * channels + c] = dv * dv * XW[(size_t)i * channels + c];
}
__global__ void edge_scatter(const int* __restrict__ src, const int* __restrict__ dst,
                             const float* __restrict__ dinv, const float* __restrict__ XW,
                             float* __restrict__ H, int channels) {
  int e = blockIdx.x, c = threadIdx.x;
  int s = src[e], d = dst[e];
  float w = dinv[s] * dinv[d];
  atomicAdd(&H[(size_t)d * channels + c], w * XW[(size_t)s * channels + c]);
}
__global__ void bias_act(float* __restrict__ H, const float* __restrict__ bias,
                         int channels, int do_relu) {
  int i = blockIdx.x, c = threadIdx.x;
  float v = H[(size_t)i * channels + c] + bias[c];
  if (do_relu) v = fmaxf(v, 0.0f);
  H[(size_t)i * channels + c] = v;
}

// ---- fused edge predictor: 16 edges/block, z_cat staged in LDS once ----
__global__ void predictor_kernel(const int* __restrict__ src, const int* __restrict__ dst,
                                 const float* __restrict__ Z,
                                 const __bf16* __restrict__ PWp1,
                                 const float* __restrict__ bp1,
                                 const float* __restrict__ Wp2,
                                 const float* __restrict__ bp2,
                                 float* __restrict__ out) {
  __shared__ float zcat[16][132];   // pad: stride 132 % 64 = 4 -> conflict-free frag reads
  __shared__ float ssum[16];
  int lane = threadIdx.x & 31;
  int wave = threadIdx.x >> 5;
  int e0   = blockIdx.x * 16;

  // cooperative coalesced gather: thread t loads one contiguous 64B segment
  {
    int m   = threadIdx.x >> 3;     // edge row 0..15
    int seg = threadIdx.x & 7;      // 8 segments of 16 floats (z_src: 0-3, z_dst: 4-7)
    int e   = e0 + m;
    const float* zrow = (seg < 4) ? (Z + (size_t)src[e] * HID + seg * 16)
                                  : (Z + (size_t)dst[e] * HID + (seg - 4) * 16);
    float4 v0 = *(const float4*)(zrow);
    float4 v1 = *(const float4*)(zrow + 4);
    float4 v2 = *(const float4*)(zrow + 8);
    float4 v3 = *(const float4*)(zrow + 12);
    float* dl = &zcat[m][seg * 16];
    *(float4*)(dl)      = v0;
    *(float4*)(dl + 4)  = v1;
    *(float4*)(dl + 8)  = v2;
    *(float4*)(dl + 12) = v3;
  }
  if (threadIdx.x < 16) ssum[threadIdx.x] = 0.0f;
  __syncthreads();

  int mrow = lane & 15;
  int koff = (lane & 16) ? 8 : 0;
  v8f acc = {};
#pragma unroll
  for (int k0 = 0; k0 < 2 * HID; k0 += 32) {
    v16bf a;
#pragma unroll
    for (int v = 0; v < 4; ++v) {
      a[2*v]     = f2bf(zcat[mrow][k0 + koff + 2*v]);
      a[2*v+1]   = f2bf(zcat[mrow][k0 + koff + 2*v + 1]);
      a[8+2*v]   = f2bf(zcat[mrow][k0 + 16 + koff + 2*v]);
      a[8+2*v+1] = f2bf(zcat[mrow][k0 + 16 + koff + 2*v + 1]);
    }
    int tile = (k0 / 32) * 4 + wave;   // Wp1 packed: 4 n-tiles
    v16bf b  = *(const v16bf*)(PWp1 + ((size_t)tile * 32 + lane) * 16);
    acc = __builtin_amdgcn_wmma_f32_16x16x32_bf16(false, a, false, b, (short)0, acc,
                                                  false, false);
  }

  // hid = relu(acc + bp1[n]); out[m] += hid * Wp2[n] (LDS reduce over n)
  int   n     = (wave * 16) + (lane & 15);
  float w2    = Wp2[n];
  float b1v   = bp1[n];
  int   mbase = (lane & 16) ? 8 : 0;
#pragma unroll
  for (int r = 0; r < 8; ++r) {
    float h = fmaxf(acc[r] + b1v, 0.0f);
    atomicAdd(&ssum[mbase + r], h * w2);
  }
  __syncthreads();
  if (threadIdx.x < 16) out[e0 + threadIdx.x] = ssum[threadIdx.x] + bp2[0];
}

extern "C" void kernel_launch(void* const* d_in, const int* in_sizes, int n_in,
                              void* d_out, int out_size, void* d_ws, size_t ws_size,
                              hipStream_t stream) {
  const float* x    = (const float*)d_in[0];
  const int*   edge = (const int*)  d_in[1];   // [2, E] row-major
  const float* W1   = (const float*)d_in[2];
  const float* b1   = (const float*)d_in[3];
  const float* W2   = (const float*)d_in[4];
  const float* b2   = (const float*)d_in[5];
  const float* Wp1  = (const float*)d_in[6];
  const float* bp1  = (const float*)d_in[7];
  const float* Wp2  = (const float*)d_in[8];
  const float* bp2  = (const float*)d_in[9];
  float* out = (float*)d_out;

  const int* src = edge;             // edge_index[0]
  const int* dst = edge + N_EDGES;   // edge_index[1]

  char*  ws  = (char*)d_ws;
  size_t off = 0;
  auto alloc = [&](size_t bytes) {
    void* p = ws + off;
    off += (bytes + 255) & ~(size_t)255;
    return p;
  };
  float*  dinv = (float*) alloc((size_t)N_NODES * 4);
  float*  XW1  = (float*) alloc((size_t)N_NODES * C1 * 4);   // reused as Z0
  float*  H1   = (float*) alloc((size_t)N_NODES * C1 * 4);
  float*  Z    = (float*) alloc((size_t)N_NODES * HID * 4);
  __bf16* pW1  = (__bf16*)alloc((size_t)(IN_C/32) * (C1/16)  * 32 * 16 * 2);
  __bf16* pW2  = (__bf16*)alloc((size_t)(C1/32)   * (HID/16) * 32 * 16 * 2);
  __bf16* pWp1 = (__bf16*)alloc((size_t)(C1/32)   * (HID/16) * 32 * 16 * 2);
  float*  Z0   = XW1;  // XW1 dead after conv1 aggregation -> reuse

  // one-time weight packing into WMMA fragment layout (bf16)
  pack_b_kernel<<<(IN_C/32) * (C1/16),  32, 0, stream>>>(W1,  pW1,  C1);
  pack_b_kernel<<<(C1/32)   * (HID/16), 32, 0, stream>>>(W2,  pW2,  HID);
  pack_b_kernel<<<(C1/32)   * (HID/16), 32, 0, stream>>>(Wp1, pWp1, HID);

  // normalization: deg^-1/2 with self-loops
  deg_init <<<(N_NODES + 255) / 256, 256, 0, stream>>>(dinv);
  deg_count<<<(N_EDGES + 255) / 256, 256, 0, stream>>>(dst, dinv);
  deg_rsqrt<<<(N_NODES + 255) / 256, 256, 0, stream>>>(dinv);

  // GCNConv 1: 256 -> 128, ReLU
  gemm_bf16_kernel<IN_C><<<N_NODES / 16, 256, 0, stream>>>(x, pW1, XW1, C1);
  selfloop_init<<<N_NODES, C1, 0, stream>>>(XW1, dinv, H1, C1);
  edge_scatter <<<N_EDGES, C1, 0, stream>>>(src, dst, dinv, XW1, H1, C1);
  bias_act     <<<N_NODES, C1, 0, stream>>>(H1, b1, C1, 1);

  // GCNConv 2: 128 -> 64
  gemm_bf16_kernel<C1><<<N_NODES / 16, 128, 0, stream>>>(H1, pW2, Z0, HID);
  selfloop_init<<<N_NODES, HID, 0, stream>>>(Z0, dinv, Z, HID);
  edge_scatter <<<N_EDGES, HID, 0, stream>>>(src, dst, dinv, Z0, Z, HID);
  bias_act     <<<N_NODES, HID, 0, stream>>>(Z, b2, HID, 0);

  // edge MLP predictor (LDS-staged gather + WMMA GEMM + ReLU + dot + bias)
  predictor_kernel<<<N_EDGES / 16, 128, 0, stream>>>(src, dst, Z, pWp1, bp1, Wp2, bp2, out);
}
